// Zencoder_59931973648924
// MI455X (gfx1250) — matrix operand order repaired and checked by reading, not compiled, not run on hardware
//
#include <hip/hip_runtime.h>
#include <hip/hip_bf16.h>

// ============================================================================
// SPADE-style encoder + masked segment-mean for MI455X (gfx1250, wave32).
// Convs = implicit GEMM on v_wmma_f32_16x16x32_f16; each wave owns a
// 16(M) x NT*16(N) strip (NT a template constant -> no EXEC predication
// around WMMA, compile-time unrolled accumulator chains).
// ============================================================================

typedef __attribute__((ext_vector_type(16))) _Float16 v16h;
typedef __attribute__((ext_vector_type(8)))  _Float16 v8h;
typedef __attribute__((ext_vector_type(8)))  float    v8f;

#define IN_EPS    1e-5f
#define NEG_SLOPE 0.2f

// ---------------------------------------------------------------------------
// Padding / layout kernels: fp32 in -> f16 NHWC padded out
// ---------------------------------------------------------------------------

__global__ void pad_reflect_nchw_to_nhwc(const float* __restrict__ in,
                                         _Float16* __restrict__ out,
                                         int B, int C, int H, int W) {
  const int Hp = H + 2, Wp = W + 2;
  const size_t n = (size_t)B * Hp * Wp * C;
  for (size_t i = (size_t)blockIdx.x * blockDim.x + threadIdx.x; i < n;
       i += (size_t)gridDim.x * blockDim.x) {
    int c = (int)(i % C);
    size_t t = i / C;
    int x = (int)(t % Wp); t /= Wp;
    int y = (int)(t % Hp);
    int b = (int)(t / Hp);
    int ys = y - 1; ys = ys < 0 ? -ys : (ys >= H ? 2 * H - 2 - ys : ys);
    int xs = x - 1; xs = xs < 0 ? -xs : (xs >= W ? 2 * W - 2 - xs : xs);
    out[i] = (_Float16)in[(((size_t)b * C + c) * H + ys) * W + xs];
  }
}

__global__ void pad_zero_nhwc(const float* __restrict__ in, _Float16* __restrict__ out,
                              int B, int H, int W, int C) {
  const int Hp = H + 2, Wp = W + 2;
  const size_t n = (size_t)B * Hp * Wp * C;
  for (size_t i = (size_t)blockIdx.x * blockDim.x + threadIdx.x; i < n;
       i += (size_t)gridDim.x * blockDim.x) {
    int c = (int)(i % C);
    size_t t = i / C;
    int x = (int)(t % Wp); t /= Wp;
    int y = (int)(t % Hp);
    int b = (int)(t / Hp);
    int ys = y - 1, xs = x - 1;
    out[i] = (ys >= 0 && ys < H && xs >= 0 && xs < W)
                 ? (_Float16)in[(((size_t)b * H + ys) * W + xs) * C + c]
                 : (_Float16)0.0f;
  }
}

__global__ void pad_reflect_nhwc(const float* __restrict__ in, _Float16* __restrict__ out,
                                 int B, int H, int W, int C) {
  const int Hp = H + 2, Wp = W + 2;
  const size_t n = (size_t)B * Hp * Wp * C;
  for (size_t i = (size_t)blockIdx.x * blockDim.x + threadIdx.x; i < n;
       i += (size_t)gridDim.x * blockDim.x) {
    int c = (int)(i % C);
    size_t t = i / C;
    int x = (int)(t % Wp); t /= Wp;
    int y = (int)(t % Hp);
    int b = (int)(t / Hp);
    int ys = y - 1; ys = ys < 0 ? -ys : (ys >= H ? 2 * H - 2 - ys : ys);
    int xs = x - 1; xs = xs < 0 ? -xs : (xs >= W ? 2 * W - 2 - xs : xs);
    out[i] = (_Float16)in[(((size_t)b * H + ys) * W + xs) * C + c];
  }
}

// ConvTranspose(k3,s2,p1,op1) lhs-dilation: fp32 [B,Hi,Wi,C] -> f16 [B,2Hi+2,2Wi+2,C]
__global__ void dilate_pad_nhwc(const float* __restrict__ in, _Float16* __restrict__ out,
                                int B, int Hi, int Wi, int C) {
  const int Hp = 2 * Hi + 2, Wp = 2 * Wi + 2;
  const int Hd = 2 * Hi - 1, Wd = 2 * Wi - 1;
  const size_t n = (size_t)B * Hp * Wp * C;
  for (size_t i = (size_t)blockIdx.x * blockDim.x + threadIdx.x; i < n;
       i += (size_t)gridDim.x * blockDim.x) {
    int c = (int)(i % C);
    size_t t = i / C;
    int x = (int)(t % Wp); t /= Wp;
    int y = (int)(t % Hp);
    int b = (int)(t / Hp);
    int yd = y - 1, xd = x - 1;
    bool v = (yd >= 0 && yd < Hd && ((yd & 1) == 0) &&
              xd >= 0 && xd < Wd && ((xd & 1) == 0));
    out[i] = v ? (_Float16)in[(((size_t)b * Hi + (yd >> 1)) * Wi + (xd >> 1)) * C + c]
               : (_Float16)0.0f;
  }
}

// weights OIHW fp32 [O,C,3,3] -> f16 Wt[o*Kp + k], k = (ky*3+kx)*C + c, tail zero
__global__ void wconvert_kernel(const float* __restrict__ w, _Float16* __restrict__ wt,
                                int O, int C, int Kp) {
  int idx = blockIdx.x * blockDim.x + threadIdx.x;
  if (idx >= O * Kp) return;
  int o = idx / Kp, k = idx - o * Kp;
  float v = 0.0f;
  if (k < 9 * C) {
    int p = k / C, c = k - p * C;               // p = ky*3+kx
    v = w[((size_t)o * C + c) * 9 + p];
  }
  wt[idx] = (_Float16)v;
}

// ---------------------------------------------------------------------------
// Implicit-GEMM conv, one wave -> 16(M) x NT*16(N) via WMMA f32_16x16x32_f16.
//   pad : NHWC f16 [B,Hp,Wp,C] (spatially padded)
//   Wt  : f16 [N, Kp], k = (ky*3+kx)*C + c (tail zero)
//   out : NHWC fp32 [B,Ho,Wo,N] = conv + bias
// A lane layout (ISA 7.12.2): row = lane&15, h = lane>>4,
//   element e -> K = 16h+e (contiguous, same as B; not the ISA table order).
// B lane layout: col = lane&15, element e -> K = 16h + e (contiguous 16xf16).
// D layout: VGPR r -> M = r + 8h, N = lane&15.
// NT is compile-time so no EXEC predication ever wraps a WMMA.
// ---------------------------------------------------------------------------
template <int NT>
__global__ __launch_bounds__(128) void wmma_conv_kernel(
    const _Float16* __restrict__ pad, const _Float16* __restrict__ Wt,
    const float* __restrict__ bias, float* __restrict__ outp,
    int B, int Hp, int Wp, int C, int Ho, int Wo, int stride, int N, int Kp) {
  const int lane = threadIdx.x & 31;
  const int wave = threadIdx.x >> 5;
  const int HoWo = Ho * Wo;
  const long M = (long)B * HoWo;
  const int mtiles = (int)(M >> 4);
  const int strips = (N >> 4) / NT;                        // exact by construction
  long tile = (long)blockIdx.x * 4 + wave;                 // wave-uniform
  if (tile >= (long)mtiles * (long)strips) return;         // whole wave exits
  const int st = (int)(tile % strips);
  const int mt = (int)(tile / strips);

  const int h   = lane >> 4;
  const int lan = lane & 15;

  // A-fragment row -> output pixel
  const int row = (mt << 4) + lan;
  const int b   = row / HoWo;
  const int r2  = row - b * HoWo;
  const int oy  = r2 / Wo;
  const int ox  = r2 - oy * Wo;
  const int iy0 = oy * stride, ix0 = ox * stride;

  // B-fragment pointers (already offset by the lane's K-half)
  const _Float16* __restrict__ wp[NT];
  int col[NT];
#pragma unroll
  for (int j = 0; j < NT; ++j) {
    col[j] = ((st * NT + j) << 4) + lan;
    wp[j] = Wt + (size_t)col[j] * Kp + (h << 4);
  }

  v8f acc[NT];
#pragma unroll
  for (int j = 0; j < NT; ++j) acc[j] = (v8f){};

  const int KC = 9 * C;

  if ((C & 31) == 0) {
    // ---- fast path: division-free hot loop ----
    const _Float16* __restrict__ rbase =
        pad + ((size_t)b * Hp + iy0) * Wp * C + (size_t)ix0 * C;
    int k0 = 0;
    for (int ky = 0; ky < 3; ++ky) {
      for (int kx = 0; kx < 3; ++kx) {
        const _Float16* __restrict__ base = rbase + ((size_t)ky * Wp + kx) * C + 16 * h;
        for (int c0 = 0; c0 < C; c0 += 32, k0 += 32) {
          const v8h g0 = *(const v8h*)(base + c0);          // K = k0+16h .. +7
          const v8h g1 = *(const v8h*)(base + c0 + 8);      // K = k0+16h+8 .. +7
          v16h a;
#pragma unroll
          for (int e = 0; e < 8; ++e) { a[e] = g0[e]; a[8 + e] = g1[e]; }
#pragma unroll
          for (int j = 0; j < NT; ++j) {
            const v16h bv = *(const v16h*)(wp[j] + k0);
            acc[j] = __builtin_amdgcn_wmma_f32_16x16x32_f16(
                false, a, false, bv, (short)0, acc[j], false, false);
          }
        }
      }
    }
  } else {
    // ---- generic path (conv0 only: C=3, KC=27, Kp=32 -> single k-step) ----
    for (int k0 = 0; k0 < Kp; k0 += 32) {
      v16h a;
#pragma unroll
      for (int e = 0; e < 16; ++e) {
        const int koff = h * 16 + e;
        const int k = k0 + koff;
        _Float16 v = (_Float16)0.0f;
        if (k < KC) {
          int p = k / C, c = k - p * C;
          int ky = p / 3, kx = p - ky * 3;
          v = pad[(((size_t)b * Hp + (iy0 + ky)) * Wp + (ix0 + kx)) * C + c];
        }
        a[e] = v;
      }
#pragma unroll
      for (int j = 0; j < NT; ++j) {
        const v16h bv = *(const v16h*)(wp[j] + k0);
        acc[j] = __builtin_amdgcn_wmma_f32_16x16x32_f16(
            false, a, false, bv, (short)0, acc[j], false, false);
      }
    }
  }

  // store D tiles (+bias)
#pragma unroll
  for (int j = 0; j < NT; ++j) {
    const float bval = bias[col[j]];
#pragma unroll
    for (int r = 0; r < 8; ++r) {
      const int orow = (mt << 4) + r + (h << 3);
      outp[(size_t)orow * N + col[j]] = acc[j][r] + bval;
    }
  }
}

// ---------------------------------------------------------------------------
// InstanceNorm (biased var) + LeakyReLU over NHWC fp32
// ---------------------------------------------------------------------------
__global__ void inorm_stats_kernel(const float* __restrict__ act, float* __restrict__ mv,
                                   int B, int HW, int C) {
  int idx = blockIdx.x * blockDim.x + threadIdx.x;
  if (idx >= B * C) return;
  int b = idx / C, c = idx - b * C;
  const float* __restrict__ p = act + (size_t)b * HW * C + c;
  float s = 0.0f, s2 = 0.0f;
  for (int i = 0; i < HW; ++i) {
    float v = p[(size_t)i * C];
    s += v; s2 += v * v;
  }
  float m = s / (float)HW;
  mv[2 * idx]     = m;
  mv[2 * idx + 1] = s2 / (float)HW - m * m;
}

// cmask = C-1, bshift = log2(HW*C)  (all powers of two here)
__global__ void inorm_apply_kernel(float* __restrict__ act, const float* __restrict__ mv,
                                   size_t n, int cmask, int bshift, int C) {
  for (size_t i = (size_t)blockIdx.x * blockDim.x + threadIdx.x; i < n;
       i += (size_t)gridDim.x * blockDim.x) {
    int c = (int)(i & (size_t)cmask);
    int b = (int)(i >> bshift);
    float m   = mv[2 * (b * C + c)];
    float var = mv[2 * (b * C + c) + 1];
    float x = (act[i] - m) * rsqrtf(var + IN_EPS);
    act[i] = x >= 0.0f ? x : NEG_SLOPE * x;
  }
}

__global__ void tanh_kernel(float* __restrict__ a, size_t n) {
  for (size_t i = (size_t)blockIdx.x * blockDim.x + threadIdx.x; i < n;
       i += (size_t)gridDim.x * blockDim.x)
    a[i] = tanhf(a[i]);
}

// ---------------------------------------------------------------------------
// Masked segment mean: block per (b,s), mask cached in LDS (16K fp32 = 64KB of
// the 320KB WGP LDS), 512 threads each reduce one feature (coalesced reads).
// ---------------------------------------------------------------------------
__global__ __launch_bounds__(512) void segmean_kernel(
    const float* __restrict__ codes,   // NHWC [B,Hc,Wc,F]
    const float* __restrict__ segmap,  // [B,S,Hs,Ws]
    const float* __restrict__ fg,      // [B,1,Hs,Ws]
    float* __restrict__ out,           // [B,S,F]
    int B, int S, int Hc, int Wc, int F, int Hs, int Ws) {
  __shared__ float seg[16384];
  __shared__ float cnt_sh;
  const int bs = blockIdx.x;
  const int b = bs / S, s = bs - b * S;
  const int t = threadIdx.x;
  const int HWc = Hc * Wc;
  const int ry = Hs / Hc, rx = Ws / Wc;   // nearest: floor(i*scale) == i*r here

  if (t == 0) cnt_sh = 0.0f;
  __syncthreads();

  float cl = 0.0f;
  for (int i = t; i < HWc; i += blockDim.x) {
    int y = i / Wc, x = i - y * Wc;
    int sy = y * ry, sx = x * rx;
    float sv = segmap[(((size_t)(b * S + s)) * Hs + sy) * Ws + sx] *
               (1.0f - fg[((size_t)b * Hs + sy) * Ws + sx]);
    seg[i] = sv;
    if (sv != 0.0f) cl += 1.0f;
  }
  atomicAdd(&cnt_sh, cl);
  __syncthreads();
  const float cnt = cnt_sh;

  for (int f = t; f < F; f += blockDim.x) {
    float accv = 0.0f;
    for (int i = 0; i < HWc; ++i) {
      float sv = seg[i];
      if (sv != 0.0f) accv += codes[((size_t)(b * HWc + i)) * F + f] * sv;
    }
    out[((size_t)(b * S + s)) * F + f] = (cnt > 0.0f) ? accv / fmaxf(cnt, 1.0f) : 0.0f;
  }
}

// ---------------------------------------------------------------------------
// Host orchestration
// ---------------------------------------------------------------------------
static inline unsigned cdiv_u(long a, long b) { return (unsigned)((a + b - 1) / b); }
static inline int ilog2(long v) { int s = 0; while ((1l << s) < v) ++s; return s; }

extern "C" void kernel_launch(void* const* d_in, const int* in_sizes, int n_in,
                              void* d_out, int out_size, void* d_ws, size_t ws_size,
                              hipStream_t stream) {
  (void)in_sizes; (void)n_in; (void)out_size; (void)ws_size;

  const float* input  = (const float*)d_in[0];   // [4,3,256,256]
  const float* segmap = (const float*)d_in[1];   // [4,20,256,256]
  const float* fg     = (const float*)d_in[2];   // [4,1,256,256]
  const float* w0 = (const float*)d_in[3];  const float* b0 = (const float*)d_in[4];
  const float* w1 = (const float*)d_in[5];  const float* b1 = (const float*)d_in[6];
  const float* w2 = (const float*)d_in[7];  const float* b2 = (const float*)d_in[8];
  const float* wtr= (const float*)d_in[9];  const float* bt = (const float*)d_in[10];
  const float* w3 = (const float*)d_in[11]; const float* b3 = (const float*)d_in[12];
  float* out = (float*)d_out;                    // [4,20,512]

  // Workspace partition (~260 MB peak):
  //  R0 fp32 acts: a0/a2/codes (max 134.3 MB)   R1 f16 padded inputs (max 34.7 MB)
  //  R2 fp32 acts: a1/a3 (max 67.2 MB)          WH f16 weights (max 2.4 MB)  MV stats
  char* ws = (char*)d_ws;
  float*    R0 = (float*)(ws);
  _Float16* R1 = (_Float16*)(ws + 0x08800000ull);
  float*    R2 = (float*)(ws + 0x0B000000ull);
  _Float16* WH = (_Float16*)(ws + 0x0F400000ull);
  float*    MV = (float*)(ws + 0x0F800000ull);

  const int B = 4;

  auto run_conv = [&](const _Float16* padp, const float* w, const float* bias,
                      float* outp, int Hp, int Wp, int C, int Ho, int Wo,
                      int stride, int N) {
    const int Kp = (9 * C + 31) & ~31;
    const long nW = (long)N * Kp;
    wconvert_kernel<<<cdiv_u(nW, 256), 256, 0, stream>>>(w, WH, N, C, Kp);
    const int ntiles = N / 16;
    const int NT = (ntiles % 4 == 0) ? 4 : 2;
    const long tiles = ((long)B * Ho * Wo / 16) * (ntiles / NT);
    if (NT == 4) {
      wmma_conv_kernel<4><<<cdiv_u(tiles, 4), 128, 0, stream>>>(
          padp, WH, bias, outp, B, Hp, Wp, C, Ho, Wo, stride, N, Kp);
    } else {
      wmma_conv_kernel<2><<<cdiv_u(tiles, 4), 128, 0, stream>>>(
          padp, WH, bias, outp, B, Hp, Wp, C, Ho, Wo, stride, N, Kp);
    }
  };
  auto run_norm = [&](float* act, int HW, int C) {
    inorm_stats_kernel<<<cdiv_u((long)B * C, 256), 256, 0, stream>>>(act, MV, B, HW, C);
    const size_t n = (size_t)B * HW * C;
    inorm_apply_kernel<<<cdiv_u((long)n, 256), 256, 0, stream>>>(
        act, MV, n, C - 1, ilog2((long)HW * C), C);
  };

  // conv0: reflect pad -> f16 [4,258,258,3]; conv 3->32 s1 -> a0 [4,256,256,32]
  pad_reflect_nchw_to_nhwc<<<cdiv_u((long)B * 258 * 258 * 3, 256), 256, 0, stream>>>(
      input, R1, B, 3, 256, 256);
  run_conv(R1, w0, b0, R0, 258, 258, 3, 256, 256, 1, 32);
  run_norm(R0, 256 * 256, 32);

  // conv1: zero pad -> f16 [4,258,258,32]; conv 32->64 s2 -> a1 [4,128,128,64]
  pad_zero_nhwc<<<cdiv_u((long)B * 258 * 258 * 32, 256), 256, 0, stream>>>(
      R0, R1, B, 256, 256, 32);
  run_conv(R1, w1, b1, R2, 258, 258, 32, 128, 128, 2, 64);
  run_norm(R2, 128 * 128, 64);

  // conv2: zero pad -> f16 [4,130,130,64]; conv 64->128 s2 -> a2 [4,64,64,128]
  pad_zero_nhwc<<<cdiv_u((long)B * 130 * 130 * 64, 256), 256, 0, stream>>>(
      R2, R1, B, 128, 128, 64);
  run_conv(R1, w2, b2, R0, 130, 130, 64, 64, 64, 2, 128);
  run_norm(R0, 64 * 64, 128);

  // convT: lhs-dilate(2)+pad(1,2) -> f16 [4,130,130,128]; conv 128->256 -> a3
  dilate_pad_nhwc<<<cdiv_u((long)B * 130 * 130 * 128, 256), 256, 0, stream>>>(
      R0, R1, B, 64, 64, 128);
  run_conv(R1, wtr, bt, R2, 130, 130, 128, 128, 128, 1, 256);
  run_norm(R2, 128 * 128, 256);

  // conv3: reflect pad -> f16 [4,130,130,256]; conv 256->512 -> codes; tanh
  pad_reflect_nhwc<<<cdiv_u((long)B * 130 * 130 * 256, 256), 256, 0, stream>>>(
      R2, R1, B, 128, 128, 256);
  run_conv(R1, w3, b3, R0, 130, 130, 256, 128, 128, 1, 512);
  tanh_kernel<<<cdiv_u((long)B * 128 * 128 * 512, 256), 256, 0, stream>>>(
      R0, (size_t)B * 128 * 128 * 512);

  // segment mean -> [4,20,512]
  segmean_kernel<<<B * 20, 512, 0, stream>>>(
      R0, segmap, fg, out, B, 20, 128, 128, 512, 256, 256);
}
